// Model_41506563948810
// MI455X (gfx1250) — matrix operand order repaired and checked by reading, not compiled, not run on hardware
//
#include <hip/hip_runtime.h>
#include <hip/hip_bf16.h>
#include <math.h>

typedef __attribute__((ext_vector_type(16))) _Float16 v16h;
typedef __attribute__((ext_vector_type(8)))  float    v8f;
typedef unsigned int u32;

#define WMMA_F16(a,b,c) __builtin_amdgcn_wmma_f32_16x16x32_f16(false,(a),false,(b),(short)0,(c),false,false)

// problem dims
constexpr int B_ = 16384, T_ = 50, D_ = 32, H_ = 128, E_ = 8, HID_ = 256, O_ = 32;

// ---- workspace layout (bytes) ----
constexpr size_t ZPACK_DW  = 32ull * 5 * 256;       // z-GEMM B-frags (Wh 4 kchunks + Wi 1) -> 40960 dwords
constexpr size_t RPACK_DW  = 5ull * 256;            // router B-frags
constexpr size_t W1PACK_DW = 8ull * 16 * 4 * 256;   // 131072
constexpr size_t W2PACK_DW = 8ull * 2 * 8 * 256;    // 32768
constexpr size_t WS_ZPACK  = 0;
constexpr size_t WS_RPACK  = WS_ZPACK + ZPACK_DW * 4;
constexpr size_t WS_W1PACK = WS_RPACK + RPACK_DW * 4;
constexpr size_t WS_W2PACK = WS_W1PACK + W1PACK_DW * 4;
constexpr size_t WS_ACC    = WS_W2PACK + W2PACK_DW * 4;   // imp[8], load[8] floats (padded to 256B)
constexpr size_t WS_GATES  = WS_ACC + 256;                // B*8 f32
constexpr size_t WS_HT     = WS_GATES + (size_t)B_ * 8 * 4; // B*128 f16
constexpr size_t WS_TOTAL  = WS_HT + (size_t)B_ * H_ * 2;

// ---- LSTM kernel LDS layout (bytes); B-frags now live in VGPRs, not LDS ----
constexpr int L_RB    = 0;            // u32[1280]   router B-frags     5120
constexpr int L_BIAS  = 5120;         // f32[512]    bi+bh              2048
constexpr int L_BXH   = 7168;         // f32[8] (padded 32)
constexpr int L_ZBUF  = 7200;         // f32[16*520]                    33280
constexpr int L_H     = 40480;        // f16[16*136]                    4352
constexpr int L_C     = 44832;        // f32[16*132]                    8448
constexpr int L_X     = 53280;        // f16[16*40]                     1280
constexpr int L_G     = 54560;        // f32[16*8]                      512
constexpr int L_GATE  = 55072;        // f32[16*8]                      512
constexpr int LSTM_SMEM = 55584;      // ~5 workgroups / WGP

// ---- experts kernel LDS layout ----
constexpr int X_H    = 0;             // f16[16*136]  4352
constexpr int X_GATE = 4352;          // f32[128]     512
constexpr int X_Y    = 4864;          // f32[16*32]   2048
constexpr int X_H1   = 6912;          // f16[8][16*264] 67584
constexpr int EXP_SMEM = 74496;

// -------- fast activations (v_exp_f32 / v_rcp_f32 paths) --------
__device__ __forceinline__ float fast_sigmoid(float x) {
  return __frcp_rn(1.f + __expf(-x));
}
__device__ __forceinline__ float fast_tanh(float x) {
  // tanh(x) = 1 - 2/(exp(2x)+1); correct limits at +/-inf
  return 1.f - 2.f * __frcp_rn(__expf(2.f * x) + 1.f);
}

// -------- fragment helpers --------
// A 16x32 f16, this file's k order (B packed to match): lane L -> row M=L&15; vgpr v, hiHalf=(L>>4):
//   k = (hi<<4) + (v<<1)  -> per lane, one contiguous 32B run:
//   bytes [row*stride*2 + kbase*2 + hi*32, +32)
__device__ __forceinline__ v16h afrag_lds(const _Float16* base, int strideH, int row, int kbase, int hi) {
  union { uint4 q[2]; v16h h; } t;
  const char* p = (const char*)base + row * strideH * 2 + kbase * 2 + hi * 32;
  t.q[0] = *(const uint4*)p;
  t.q[1] = *(const uint4*)(p + 16);
  return t.h;
}

// B 32x16 f16, same k order as A: lane 0-15 -> N=lane, K=0..15; lane 16-31 -> N=lane-16, K=16..31.
// Packed blob: per frag [32 lanes][8 dwords] contiguous per lane -> 2x b128 per lane.
__device__ __forceinline__ v16h bfrag_lds(const u32* blob, int frag, int lane) {
  union { uint4 q[2]; v16h h; } t;
  const uint4* p = (const uint4*)(blob + frag * 256 + lane * 8);
  t.q[0] = p[0];
  t.q[1] = p[1];
  return t.h;
}

__device__ __forceinline__ v16h bfrag_g(const uint4* blob, int frag, int lane) {
  union { uint4 q[2]; v16h h; } t;
  const uint4* p = blob + frag * 64 + lane * 2;
  t.q[0] = p[0];
  t.q[1] = p[1];
  return t.h;
}

__device__ __forceinline__ v8f vzero8() {
  v8f a;
#pragma unroll
  for (int r = 0; r < 8; ++r) a[r] = 0.f;
  return a;
}

__device__ __forceinline__ u32 pack2h(float f0, float f1) {
  union { _Float16 h[2]; u32 u; } t;
  t.h[0] = (_Float16)f0; t.h[1] = (_Float16)f1;
  return t.u;
}

// ================= pack kernels =================
__global__ void pack_z_kernel(const float* __restrict__ Wi, const float* __restrict__ Wh, u32* __restrict__ dst) {
  int idx = blockIdx.x * 256 + threadIdx.x;
  if (idx >= (int)ZPACK_DW) return;
  int nt = idx / 1280, rem = idx % 1280;
  int kc = rem / 256, li = rem % 256;
  int lane = li / 8, d = li % 8;
  int n = nt * 16 + (lane & 15);
  int k0 = (lane >> 4) * 16 + 2 * d;
  float f0, f1;
  if (kc < 4) { int kb = kc * 32; f0 = Wh[(kb + k0) * 512 + n]; f1 = Wh[(kb + k0 + 1) * 512 + n]; }
  else        { f0 = Wi[k0 * 512 + n]; f1 = Wi[(k0 + 1) * 512 + n]; }
  dst[idx] = pack2h(f0, f1);
}

__global__ void pack_router_kernel(const float* __restrict__ Wxh, const float* __restrict__ Whh, u32* __restrict__ dst) {
  int idx = blockIdx.x * 256 + threadIdx.x;
  if (idx >= (int)RPACK_DW) return;
  int kc = idx / 256, li = idx % 256;
  int lane = li / 8, d = li % 8;
  int n = lane & 15;
  int k0 = (lane >> 4) * 16 + 2 * d;
  float f0 = 0.f, f1 = 0.f;
  if (n < 8) {
    if (kc < 4) { f0 = Wxh[(kc * 32 + k0) * 8 + n]; f1 = Wxh[(kc * 32 + k0 + 1) * 8 + n]; }
    else        { if (k0 < 8) f0 = Whh[k0 * 8 + n]; if (k0 + 1 < 8) f1 = Whh[(k0 + 1) * 8 + n]; }
  }
  dst[idx] = pack2h(f0, f1);
}

__global__ void pack_w1_kernel(const float* __restrict__ W1, u32* __restrict__ dst) {
  int idx = blockIdx.x * 256 + threadIdx.x;
  if (idx >= (int)W1PACK_DW) return;
  int e = idx / 16384, rem = idx % 16384;
  int nt = rem / 1024, kc = (rem / 256) % 4, li = rem % 256;
  int lane = li / 8, d = li % 8;
  int n = nt * 16 + (lane & 15);
  int k0 = (lane >> 4) * 16 + 2 * d;
  float f0 = W1[(size_t)(e * 128 + kc * 32 + k0) * 256 + n];
  float f1 = W1[(size_t)(e * 128 + kc * 32 + k0 + 1) * 256 + n];
  dst[idx] = pack2h(f0, f1);
}

__global__ void pack_w2_kernel(const float* __restrict__ W2, u32* __restrict__ dst) {
  int idx = blockIdx.x * 256 + threadIdx.x;
  if (idx >= (int)W2PACK_DW) return;
  int e = idx / 4096;
  int nt = (idx / 2048) % 2, kc = (idx / 256) % 8, li = idx % 256;
  int lane = li / 8, d = li % 8;
  int n = nt * 16 + (lane & 15);
  int k0 = (lane >> 4) * 16 + 2 * d;
  float f0 = W2[(size_t)(e * 256 + kc * 32 + k0) * 32 + n];
  float f1 = W2[(size_t)(e * 256 + kc * 32 + k0 + 1) * 32 + n];
  dst[idx] = pack2h(f0, f1);
}

__global__ void zero_acc_kernel(float* __restrict__ acc) {
  int i = threadIdx.x;
  if (i < 16) acc[i] = 0.f;
}

// ================= fused LSTM + router + gating =================
__global__ void __launch_bounds__(256) lstm_router_kernel(
    const float* __restrict__ x, const float* __restrict__ bi, const float* __restrict__ bh,
    const float* __restrict__ bxh,
    const u32* __restrict__ zpack, const u32* __restrict__ rpack,
    float* __restrict__ gatesG, _Float16* __restrict__ hTG,
    float* __restrict__ impAcc, float* __restrict__ loadAcc) {
  extern __shared__ char smem[];
  u32*      smR    = (u32*)(smem + L_RB);
  float*    smBias = (float*)(smem + L_BIAS);
  float*    smBxh  = (float*)(smem + L_BXH);
  float*    smZb   = (float*)(smem + L_ZBUF);
  _Float16* smH    = (_Float16*)(smem + L_H);
  float*    smC    = (float*)(smem + L_C);
  _Float16* smX    = (_Float16*)(smem + L_X);
  float*    smG    = (float*)(smem + L_G);
  float*    smGate = (float*)(smem + L_GATE);

  const int tid  = threadIdx.x;
  const int lane = tid & 31;
  const int w    = tid >> 5;
  const int hi   = (lane >> 4) & 1;
  const int arow = lane & 15;
  const int b0   = blockIdx.x * 16;

  // router frags to LDS (small); z-GEMM B-frags -> registers for entire T loop
  {
    const uint4* s2 = (const uint4*)rpack; uint4* d2 = (uint4*)smR;
    for (int i = tid; i < (int)(RPACK_DW / 4); i += 256) d2[i] = s2[i];
  }
  for (int i = tid; i < 512; i += 256) smBias[i] = bi[i] + bh[i];
  if (tid < 8) smBxh[tid] = bxh[tid];
  for (int i = tid; i < 16 * 136; i += 256) smH[i] = (_Float16)0.f;
  for (int i = tid; i < 16 * 132; i += 256) smC[i] = 0.f;
  for (int i = tid; i < 128; i += 256) smG[i] = 0.f;

  // hold this wave's 20 B fragments (4 N-tiles x 5 k-chunks) in VGPRs
  v16h bz[5][4];
#pragma unroll
  for (int kc = 0; kc < 5; ++kc)
#pragma unroll
    for (int i = 0; i < 4; ++i)
      bz[kc][i] = bfrag_g((const uint4*)zpack, (w * 4 + i) * 5 + kc, lane);

  __syncthreads();

  for (int t = 0; t < T_; ++t) {
    // stage x_t (f32 -> f16)
    for (int i = tid; i < 16 * D_; i += 256) {
      int row = i >> 5, j = i & 31;
      smX[row * 40 + j] = (_Float16)x[((size_t)(b0 + row) * T_ + t) * D_ + j];
    }
    __syncthreads();

    // z[16,512] = h@Wh + x@Wi ; wave w owns N-tiles 4w..4w+3.
    // B operands are register-resident; only A frags come from LDS (double-buffered).
    v8f acc[4];
#pragma unroll
    for (int i = 0; i < 4; ++i) acc[i] = vzero8();

    v16h aCur = afrag_lds(smH, 136, arow, 0, hi);
#pragma unroll
    for (int kc = 0; kc < 5; ++kc) {
      v16h aNxt;
      if (kc < 4) {
        aNxt = (kc + 1 < 4) ? afrag_lds(smH, 136, arow, (kc + 1) * 32, hi)
                            : afrag_lds(smX, 40, arow, 0, hi);
      }
#pragma unroll
      for (int i = 0; i < 4; ++i) acc[i] = WMMA_F16(aCur, bz[kc][i], acc[i]);
      if (kc < 4) aCur = aNxt;
    }

#pragma unroll
    for (int i = 0; i < 4; ++i) {
      int col = (w * 4 + i) * 16 + arow;
#pragma unroll
      for (int r = 0; r < 8; ++r) smZb[(r + hi * 8) * 520 + col] = acc[i][r];
    }
    __syncthreads();

    // elementwise LSTM cell (i,f,g,o gate order)
#pragma unroll
    for (int ii = 0; ii < 8; ++ii) {
      int e2 = tid + ii * 256;
      int row = e2 >> 7, j = e2 & 127;
      float zi = smZb[row * 520 + j]       + smBias[j];
      float zf = smZb[row * 520 + 128 + j] + smBias[128 + j];
      float zg = smZb[row * 520 + 256 + j] + smBias[256 + j];
      float zo = smZb[row * 520 + 384 + j] + smBias[384 + j];
      float ig = fast_sigmoid(zi);
      float fg = fast_sigmoid(zf);
      float gg = fast_tanh(zg);
      float og = fast_sigmoid(zo);
      float c  = fg * smC[row * 132 + j] + ig * gg;
      smC[row * 132 + j] = c;
      smH[row * 136 + j] = (_Float16)(og * fast_tanh(c));
    }
    __syncthreads();

    // router recurrence: g = h@Wxh + bxh + g@Whh  (wave 0, WMMA on padded E=16)
    if (w == 0) {
      v8f g = vzero8();
#pragma unroll
      for (int kc = 0; kc < 4; ++kc) {
        v16h a = afrag_lds(smH, 136, arow, kc * 32, hi);
        v16h b = bfrag_lds(smR, kc, lane);
        g = WMMA_F16(a, b, g);
      }
      {
        v16h a;
#pragma unroll
        for (int d = 0; d < 8; ++d) {
          int k = (hi << 4) + (d << 1);
          a[2 * d]     = (k < 8)     ? (_Float16)smG[arow * 8 + k]     : (_Float16)0.f;
          a[2 * d + 1] = (k + 1 < 8) ? (_Float16)smG[arow * 8 + k + 1] : (_Float16)0.f;
        }
        v16h b = bfrag_lds(smR, 4, lane);
        g = WMMA_F16(a, b, g);
      }
      if (arow < 8) {
#pragma unroll
        for (int r = 0; r < 8; ++r) smG[(r + hi * 8) * 8 + arow] = g[r] + smBxh[arow];
      }
    }
    __syncthreads();
  }

  // top-2 gating + softmax per row
  if (tid < 16) {
    int row = tid;
    float best = -1e30f; int i0 = 0;
    for (int e = 0; e < 8; ++e) { float v = smG[row * 8 + e]; if (v > best) { best = v; i0 = e; } }
    float best2 = -1e30f; int i1 = 0;
    for (int e = 0; e < 8; ++e) { if (e == i0) continue; float v = smG[row * 8 + e]; if (v > best2) { best2 = v; i1 = e; } }
    float e1 = __expf(best2 - best);
    float s  = 1.f + e1;
    for (int e = 0; e < 8; ++e) smGate[row * 8 + e] = 0.f;
    smGate[row * 8 + i0] = 1.f / s;
    smGate[row * 8 + i1] = e1 / s;
  }
  __syncthreads();

  for (int i = tid; i < 128; i += 256) gatesG[(size_t)b0 * 8 + i] = smGate[i];
  for (int i = tid; i < 16 * 128; i += 256) {
    int row = i >> 7, j = i & 127;
    hTG[(size_t)(b0 + row) * 128 + j] = smH[row * 136 + j];
  }
  if (tid < 8) {
    float imp = 0.f, ld = 0.f;
    for (int r = 0; r < 16; ++r) { float v = smGate[r * 8 + tid]; imp += v; if (v > 0.f) ld += 1.f; }
    atomicAdd(&impAcc[tid], imp);
    atomicAdd(&loadAcc[tid], ld);
  }
}

// ================= experts (one expert per wave) =================
__global__ void __launch_bounds__(256) experts_kernel(
    const _Float16* __restrict__ hT, const float* __restrict__ gatesG,
    const u32* __restrict__ w1p, const u32* __restrict__ w2p,
    const float* __restrict__ b1, const float* __restrict__ b2,
    float* __restrict__ out) {
  extern __shared__ char smem[];
  _Float16* smH    = (_Float16*)(smem + X_H);
  float*    smGate = (float*)(smem + X_GATE);
  float*    smY    = (float*)(smem + X_Y);
  _Float16* smH1   = (_Float16*)(smem + X_H1);

  const int tid = threadIdx.x, lane = tid & 31, w = tid >> 5;
  const int hi = (lane >> 4) & 1, arow = lane & 15;
  const int b0 = blockIdx.x * 16;

  for (int i = tid; i < 16 * 128; i += 256) {
    int row = i >> 7, j = i & 127;
    smH[row * 136 + j] = hT[(size_t)(b0 + row) * 128 + j];
  }
  for (int i = tid; i < 128; i += 256) smGate[i] = gatesG[(size_t)b0 * 8 + i];
  for (int i = tid; i < 512; i += 256) smY[i] = 0.f;
  __syncthreads();

  const int e = w;
  _Float16* myH1 = smH1 + w * (16 * 264);

  v16h ha[4];
#pragma unroll
  for (int kc = 0; kc < 4; ++kc) ha[kc] = afrag_lds(smH, 136, arow, kc * 32, hi);

  const uint4* w1q = (const uint4*)w1p;
  const uint4* w2q = (const uint4*)w2p;

  // h1 = relu(hT @ W1[e] + b1[e])  -> per-wave f16 LDS
#pragma unroll 4
  for (int nt = 0; nt < 16; ++nt) {
    v8f acc = vzero8();
#pragma unroll
    for (int kc = 0; kc < 4; ++kc) {
      v16h b = bfrag_g(w1q, (e * 16 + nt) * 4 + kc, lane);
      acc = WMMA_F16(ha[kc], b, acc);
    }
    int col = nt * 16 + arow;
#pragma unroll
    for (int r = 0; r < 8; ++r) {
      float v = acc[r] + b1[e * 256 + col];
      myH1[(r + hi * 8) * 264 + col] = (_Float16)(v > 0.f ? v : 0.f);
    }
  }

  // eo = h1 @ W2[e] + b2[e]; combine with gates into shared y
#pragma unroll
  for (int nt = 0; nt < 2; ++nt) {
    v8f acc = vzero8();
#pragma unroll
    for (int kc = 0; kc < 8; ++kc) {
      v16h a = afrag_lds(myH1, 264, arow, kc * 32, hi);
      v16h b = bfrag_g(w2q, (e * 2 + nt) * 8 + kc, lane);
      acc = WMMA_F16(a, b, acc);
    }
    int col = nt * 16 + arow;
#pragma unroll
    for (int r = 0; r < 8; ++r) {
      int row = r + hi * 8;
      float v  = acc[r] + b2[e * 32 + col];
      float gt = smGate[row * 8 + e];
      atomicAdd(&smY[row * 32 + col], gt * v);
    }
  }
  __syncthreads();

  for (int i = tid; i < 512; i += 256) {
    int row = i >> 5, col = i & 31;
    out[(size_t)(b0 + row) * 32 + col] = smY[row * 32 + col];
  }
}

// ================= loss =================
__global__ void loss_kernel(const float* __restrict__ acc, float* __restrict__ out) {
  if (threadIdx.x == 0 && blockIdx.x == 0) {
    float loss = 0.f;
    for (int which = 0; which < 2; ++which) {
      const float* v = acc + which * 8;
      float m = 0.f;
      for (int e = 0; e < 8; ++e) m += v[e];
      m /= 8.f;
      float var = 0.f;
      for (int e = 0; e < 8; ++e) { float d = v[e] - m; var += d * d; }
      var /= 7.f;  // ddof=1
      loss += 0.1f * (var / (m * m + 1e-10f));
    }
    out[(size_t)B_ * O_] = loss;
  }
}

extern "C" void kernel_launch(void* const* d_in, const int* in_sizes, int n_in,
                              void* d_out, int out_size, void* d_ws, size_t ws_size,
                              hipStream_t stream) {
  const float* x   = (const float*)d_in[0];
  const float* Wi  = (const float*)d_in[1];
  const float* Wh  = (const float*)d_in[2];
  const float* bi  = (const float*)d_in[3];
  const float* bh  = (const float*)d_in[4];
  const float* Wxh = (const float*)d_in[5];
  const float* bxh = (const float*)d_in[6];
  const float* Whh = (const float*)d_in[7];
  const float* W1  = (const float*)d_in[8];
  const float* b1  = (const float*)d_in[9];
  const float* W2  = (const float*)d_in[10];
  const float* b2  = (const float*)d_in[11];
  float* out = (float*)d_out;
  char* ws = (char*)d_ws;
  (void)in_sizes; (void)n_in; (void)out_size; (void)ws_size;

  u32*      zp   = (u32*)(ws + WS_ZPACK);
  u32*      rp   = (u32*)(ws + WS_RPACK);
  u32*      w1p  = (u32*)(ws + WS_W1PACK);
  u32*      w2p  = (u32*)(ws + WS_W2PACK);
  float*    accp = (float*)(ws + WS_ACC);
  float*    gat  = (float*)(ws + WS_GATES);
  _Float16* hTp  = (_Float16*)(ws + WS_HT);

  pack_z_kernel<<<(int)((ZPACK_DW + 255) / 256), 256, 0, stream>>>(Wi, Wh, zp);
  pack_router_kernel<<<(int)((RPACK_DW + 255) / 256), 256, 0, stream>>>(Wxh, Whh, rp);
  pack_w1_kernel<<<(int)((W1PACK_DW + 255) / 256), 256, 0, stream>>>(W1, w1p);
  pack_w2_kernel<<<(int)((W2PACK_DW + 255) / 256), 256, 0, stream>>>(W2, w2p);
  zero_acc_kernel<<<1, 32, 0, stream>>>(accp);

  lstm_router_kernel<<<B_ / 16, 256, LSTM_SMEM, stream>>>(
      x, bi, bh, bxh, zp, rp, gat, hTp, accp, accp + 8);

  experts_kernel<<<B_ / 16, 256, EXP_SMEM, stream>>>(
      hTp, gat, w1p, w2p, b1, b2, out);

  loss_kernel<<<1, 32, 0, stream>>>(accp, out);
}
